// FastAttention_62749472195327
// MI455X (gfx1250) — compile-verified, matrix operand order checked by reading, not run on hardware
//
#include <hip/hip_runtime.h>
#include <hip/hip_bf16.h>

// ---------------------------------------------------------------------------
// FAVOR+ causal linear attention, fused, CDNA5 (gfx1250) wave32 + WMMA bf16.
// b=4 h=16 n=4096 d=64 m=256 chunk=64.
// Round 3: the global-max pre-pass now uses the same WMMA projection path as
// the main kernel (no more scalar FMAC dot products, and the max is computed
// in arithmetic identical to the main kernel's dd_k).
// ---------------------------------------------------------------------------

typedef __attribute__((ext_vector_type(16))) __bf16 v16bf;
typedef __attribute__((ext_vector_type(8)))  float  v8f;

#define DNORM 0.35355339059327373f   // 64^-0.25
#define RATIO 0.0625f                // 256^-0.5
#define KEPS  1e-4f

#define MPAD 264   // qp/kp/KVT leading dim over m (256 + 8)
#define SPAD 72    // stage / A leading dim (64 + 8)
#define SMEM_BYTES 157440

__device__ __forceinline__ v8f wmma_bf16(v16bf a, v16bf b, v8f c) {
  return __builtin_amdgcn_wmma_f32_16x16x32_bf16(false, a, false, b, (short)0, c,
                                                 false, false);
}

// A/B fragment, source row-major with K contiguous along a row.
// element (rowDim = row0 + laneLo, K = k0 + laneHi*8 + g*16 + vv*2 (+1))
__device__ __forceinline__ v16bf frag_rowK_bf(const __bf16* base, int row0, int k0,
                                              int ld, int laneLo, int laneHi) {
  v16bf a;
  const __bf16* p = base + (row0 + laneLo) * ld + k0 + laneHi * 8;
#pragma unroll
  for (int g = 0; g < 2; ++g)
#pragma unroll
    for (int vv = 0; vv < 4; ++vv) {
      int kk = g * 16 + vv * 2;
      a[g * 8 + vv * 2]     = p[kk];
      a[g * 8 + vv * 2 + 1] = p[kk + 1];
    }
  return a;
}

__device__ __forceinline__ v16bf frag_rowK_f32(const float* base, int row0, int k0,
                                               int ld, int laneLo, int laneHi) {
  v16bf a;
  const float* p = base + (size_t)(row0 + laneLo) * ld + k0 + laneHi * 8;
#pragma unroll
  for (int g = 0; g < 2; ++g)
#pragma unroll
    for (int vv = 0; vv < 4; ++vv) {
      int kk = g * 16 + vv * 2;
      a[g * 8 + vv * 2]     = (__bf16)p[kk];
      a[g * 8 + vv * 2 + 1] = (__bf16)p[kk + 1];
    }
  return a;
}

// Fragment where K runs down the rows of the source (used only for kpT in the
// KV-state update; everything else is contiguous-K).
__device__ __forceinline__ v16bf frag_colK_bf(const __bf16* base, int k0, int col0,
                                              int ld, int laneLo, int laneHi) {
  v16bf a;
  const __bf16* p = base + (k0 + laneHi * 8) * ld + col0 + laneLo;
#pragma unroll
  for (int g = 0; g < 2; ++g)
#pragma unroll
    for (int vv = 0; vv < 4; ++vv) {
      int kk = g * 16 + vv * 2;
      a[g * 8 + vv * 2]     = p[kk * ld];
      a[g * 8 + vv * 2 + 1] = p[(kk + 1) * ld];
    }
  return a;
}

// ---------------------------------------------------------------------------
// Phase 1: global max of k's data_dash = (k * d^-0.25) @ projT, per-block max,
// computed with the same bf16 WMMA path the main kernel uses for dd_k.
// ---------------------------------------------------------------------------
__global__ __launch_bounds__(256) void kmax_partial(const float* __restrict__ kg,
                                                    const float* __restrict__ proj,
                                                    float* __restrict__ part) {
  __shared__ __bf16 stg[64 * SPAD];
  __shared__ float  sred[8];
  const int tid    = threadIdx.x;
  const int w      = tid >> 5;
  const int lane   = tid & 31;
  const int laneLo = lane & 15;
  const int laneHi = lane >> 4;
  const int bx     = blockIdx.x;         // 0..4095 = (bh*64 + chunk)
  const int bh     = bx >> 6, ch = bx & 63;
  const float* kbase = kg + ((size_t)bh * 4096 + (size_t)ch * 64) * 64;

  // proj B-fragments for this wave's 32-feature slice (K = d, 2 tiles).
  v16bf PB[2][2];
#pragma unroll
  for (int kt = 0; kt < 2; ++kt)
#pragma unroll
    for (int nt = 0; nt < 2; ++nt)
      PB[kt][nt] = frag_rowK_f32(proj, w * 32 + nt * 16, kt * 32, 64, laneLo, laneHi);

  for (int i = tid; i < 4096; i += 256)
    stg[(i >> 6) * SPAD + (i & 63)] = (__bf16)(kbase[i] * DNORM);
  __syncthreads();

  v8f dd[4][2];
#pragma unroll
  for (int mt = 0; mt < 4; ++mt)
#pragma unroll
    for (int nt = 0; nt < 2; ++nt)
#pragma unroll
      for (int r = 0; r < 8; ++r) dd[mt][nt][r] = 0.0f;
#pragma unroll
  for (int kt = 0; kt < 2; ++kt)
#pragma unroll
    for (int mt = 0; mt < 4; ++mt) {
      v16bf a = frag_rowK_bf(stg, mt * 16, kt * 32, SPAD, laneLo, laneHi);
#pragma unroll
      for (int nt = 0; nt < 2; ++nt) dd[mt][nt] = wmma_bf16(a, PB[kt][nt], dd[mt][nt]);
    }

  float mx = -3.4e38f;
#pragma unroll
  for (int mt = 0; mt < 4; ++mt)
#pragma unroll
    for (int nt = 0; nt < 2; ++nt)
#pragma unroll
      for (int r = 0; r < 8; ++r) mx = fmaxf(mx, dd[mt][nt][r]);
  for (int off = 1; off < 32; off <<= 1) mx = fmaxf(mx, __shfl_xor(mx, off, 32));
  if (lane == 0) sred[w] = mx;
  __syncthreads();
  if (tid == 0) {
    float m = sred[0];
#pragma unroll
    for (int j = 1; j < 8; ++j) m = fmaxf(m, sred[j]);
    part[bx] = m;
  }
}

__global__ __launch_bounds__(256) void kmax_reduce(const float* __restrict__ part,
                                                   float* __restrict__ out) {
  __shared__ float s[256];
  const int tid = threadIdx.x;
  float m = -3.4e38f;
  for (int i = tid; i < 4096; i += 256) m = fmaxf(m, part[i]);
  s[tid] = m;
  __syncthreads();
  for (int st = 128; st > 0; st >>= 1) {
    if (tid < st) s[tid] = fmaxf(s[tid], s[tid + st]);
    __syncthreads();
  }
  if (tid == 0) out[0] = s[0];
}

// ---------------------------------------------------------------------------
// Main fused kernel: one block per (b,h); sequential scan over 64 chunks.
// ---------------------------------------------------------------------------
__global__ __launch_bounds__(256) void perf_causal_main(
    const float* __restrict__ qg, const float* __restrict__ kg,
    const float* __restrict__ vg, const float* __restrict__ proj,
    const float* __restrict__ mkp, float* __restrict__ outg) {
  extern __shared__ char smem[];
  __bf16* sh_qp  = (__bf16*)(smem);            // [64][MPAD] bf16  33792 B
  __bf16* sh_kp  = (__bf16*)(smem + 33792);    // [64][MPAD] bf16  33792 B
  float*  sh_KVT = (float*) (smem + 67584);    // [64(e)][MPAD(m)] 67584 B (transposed state)
  __bf16* stage  = (__bf16*)(smem + 135168);   // [64][SPAD] bf16   9216 B
  __bf16* sh_A   = (__bf16*)(smem + 144384);   // [64][SPAD] bf16   9216 B
  float*  red    = (float*) (smem + 153600);   // [64][8]           2048 B
  float*  rowmx  = (float*) (smem + 155648);   // [64]
  float*  diag   = (float*) (smem + 155904);   // [64]
  float*  denomv = (float*) (smem + 156160);   // [64]
  float*  kstate = (float*) (smem + 156416);   // [256] -> end 157440

  const int tid    = threadIdx.x;
  const int w      = tid >> 5;          // wave 0..7
  const int lane   = tid & 31;
  const int laneLo = lane & 15;
  const int laneHi = lane >> 4;
  const int bh     = blockIdx.x;        // 0..63
  const float Mk   = mkp[0];

  // Persistent proj B-fragments: this wave's 32-feature slice, K=d (2 tiles).
  v16bf PB[2][2];
#pragma unroll
  for (int kt = 0; kt < 2; ++kt)
#pragma unroll
    for (int nt = 0; nt < 2; ++nt)
      PB[kt][nt] = frag_rowK_f32(proj, w * 32 + nt * 16, kt * 32, 64, laneLo, laneHi);

  // Zero scan state.
  for (int i = tid; i < 64 * MPAD; i += 256) sh_KVT[i] = 0.0f;
  kstate[tid] = 0.0f;
  __syncthreads();

  const size_t base = (size_t)bh * 4096 * 64;

  for (int ch = 0; ch < 64; ++ch) {
    const size_t cb = base + (size_t)ch * 64 * 64;

    // ---- prefetch next chunk into cache (global_prefetch_b8) ------------
    if (ch + 1 < 64) {
      const size_t nb = cb + 4096;
      __builtin_prefetch(qg + nb + (size_t)tid * 16, 0, 1);
      __builtin_prefetch(kg + nb + (size_t)tid * 16, 0, 1);
      __builtin_prefetch(vg + nb + (size_t)tid * 16, 0, 1);
    }

    // ---- stage q (pre-scaled) + diag_q ----------------------------------
    for (int i = tid; i < 4096; i += 256)
      stage[(i >> 6) * SPAD + (i & 63)] = (__bf16)(qg[cb + i] * DNORM);
    if (tid < 64) {
      const float* row = qg + cb + (size_t)tid * 64;
      float s = 0.f;
#pragma unroll 8
      for (int d = 0; d < 64; ++d) s += row[d] * row[d];
      diag[tid] = 0.0625f * s;   // 0.5 * d^-0.5 * sum(q^2)
    }
    __syncthreads();

    // ---- dd_q = qn @ projT (per-wave 64x32 slice) -----------------------
    v8f dd[4][2];
#pragma unroll
    for (int mt = 0; mt < 4; ++mt)
#pragma unroll
      for (int nt = 0; nt < 2; ++nt)
#pragma unroll
        for (int r = 0; r < 8; ++r) dd[mt][nt][r] = 0.0f;
#pragma unroll
    for (int kt = 0; kt < 2; ++kt)
#pragma unroll
      for (int mt = 0; mt < 4; ++mt) {
        v16bf a = frag_rowK_bf(stage, mt * 16, kt * 32, SPAD, laneLo, laneHi);
#pragma unroll
        for (int nt = 0; nt < 2; ++nt) dd[mt][nt] = wmma_bf16(a, PB[kt][nt], dd[mt][nt]);
      }
    // per-wave row max over this feature slice
#pragma unroll
    for (int mt = 0; mt < 4; ++mt)
#pragma unroll
      for (int r = 0; r < 8; ++r) {
        float v = fmaxf(dd[mt][0][r], dd[mt][1][r]);
        for (int off = 1; off < 16; off <<= 1) v = fmaxf(v, __shfl_xor(v, off, 16));
        if (laneLo == 0) red[(mt * 16 + laneHi * 8 + r) * 8 + w] = v;
      }
    __syncthreads();
    if (tid < 64) {
      float m = red[tid * 8];
#pragma unroll
      for (int j = 1; j < 8; ++j) m = fmaxf(m, red[tid * 8 + j]);
      rowmx[tid] = m;
    }
    __syncthreads();
    // qp = ratio*(exp(dd - diag - rowmax) + eps)
#pragma unroll
    for (int mt = 0; mt < 4; ++mt)
#pragma unroll
      for (int nt = 0; nt < 2; ++nt)
#pragma unroll
        for (int r = 0; r < 8; ++r) {
          int row = mt * 16 + laneHi * 8 + r;
          int col = w * 32 + nt * 16 + laneLo;
          float x = dd[mt][nt][r] - diag[row] - rowmx[row];
          sh_qp[row * MPAD + col] = (__bf16)(RATIO * (__expf(x) + KEPS));
        }
    __syncthreads();

    // ---- stage k + diag_k ----------------------------------------------
    for (int i = tid; i < 4096; i += 256)
      stage[(i >> 6) * SPAD + (i & 63)] = (__bf16)(kg[cb + i] * DNORM);
    if (tid < 64) {
      const float* row = kg + cb + (size_t)tid * 64;
      float s = 0.f;
#pragma unroll 8
      for (int d = 0; d < 64; ++d) s += row[d] * row[d];
      diag[tid] = 0.0625f * s;
    }
    __syncthreads();

    // ---- dd_k, kp (global max Mk); keep per-column partials in regs -----
#pragma unroll
    for (int mt = 0; mt < 4; ++mt)
#pragma unroll
      for (int nt = 0; nt < 2; ++nt)
#pragma unroll
        for (int r = 0; r < 8; ++r) dd[mt][nt][r] = 0.0f;
#pragma unroll
    for (int kt = 0; kt < 2; ++kt)
#pragma unroll
      for (int mt = 0; mt < 4; ++mt) {
        v16bf a = frag_rowK_bf(stage, mt * 16, kt * 32, SPAD, laneLo, laneHi);
#pragma unroll
        for (int nt = 0; nt < 2; ++nt) dd[mt][nt] = wmma_bf16(a, PB[kt][nt], dd[mt][nt]);
      }
    float csum0 = 0.f, csum1 = 0.f;   // colsum(kp) partials for this lane's cols
#pragma unroll
    for (int mt = 0; mt < 4; ++mt)
#pragma unroll
      for (int nt = 0; nt < 2; ++nt)
#pragma unroll
        for (int r = 0; r < 8; ++r) {
          int row = mt * 16 + laneHi * 8 + r;
          int col = w * 32 + nt * 16 + laneLo;
          float x  = dd[mt][nt][r] - diag[row] - Mk;
          float pv = RATIO * (__expf(x) + KEPS);
          sh_kp[row * MPAD + col] = (__bf16)pv;
          if (nt == 0) csum0 += pv; else csum1 += pv;
        }
    __syncthreads();

    // ---- stage v TRANSPOSED: stage[e][c] --------------------------------
    for (int i = tid; i < 4096; i += 256)
      stage[(i & 63) * SPAD + (i >> 6)] = (__bf16)(vg[cb + i]);

    // ---- A = tril(qp @ kpT), masked, -> sh_A ----------------------------
    {
      int mtA = w >> 1;
#pragma unroll
      for (int tt = 0; tt < 2; ++tt) {
        int ntA = (w & 1) * 2 + tt;
        v8f a;
#pragma unroll
        for (int r = 0; r < 8; ++r) a[r] = 0.0f;
#pragma unroll
        for (int kt = 0; kt < 8; ++kt) {
          v16bf af = frag_rowK_bf(sh_qp, mtA * 16, kt * 32, MPAD, laneLo, laneHi);
          v16bf bf = frag_rowK_bf(sh_kp, ntA * 16, kt * 32, MPAD, laneLo, laneHi);
          a = wmma_bf16(af, bf, a);
        }
#pragma unroll
        for (int r = 0; r < 8; ++r) {
          int row = mtA * 16 + laneHi * 8 + r;
          int col = ntA * 16 + laneLo;
          sh_A[row * SPAD + col] = (__bf16)((col <= row) ? a[r] : 0.0f);
        }
      }
    }
    __syncthreads();

    // ---- denominator: 4 threads per row, width-4 shuffle reduce ---------
    {
      int row = tid >> 2, sub = tid & 3;
      float s = 0.f;
      const __bf16* ar = sh_A + row * SPAD + sub * 16;
#pragma unroll
      for (int c = 0; c < 16; ++c) s += (float)ar[c];
      const __bf16* qr = sh_qp + row * MPAD + sub * 64;
      const float*  ks = kstate + sub * 64;
#pragma unroll 8
      for (int m = 0; m < 64; ++m) s += (float)qr[m] * ks[m];
      s += __shfl_xor(s, 1, 4);
      s += __shfl_xor(s, 2, 4);
      if (sub == 0) denomv[row] = 1.0f / s;
    }
    // ---- out = qp@KV + tril(A)@v (per-wave 16x32 of the 64x64 output) ---
    v8f accO[2];
    const int mtO = w >> 1;
#pragma unroll
    for (int tt = 0; tt < 2; ++tt) {
      int et = (w & 1) * 2 + tt;
      v8f acc;
#pragma unroll
      for (int r = 0; r < 8; ++r) acc[r] = 0.0f;
#pragma unroll
      for (int kt = 0; kt < 8; ++kt) {   // inter-chunk: K over m=256
        v16bf a = frag_rowK_bf(sh_qp, mtO * 16, kt * 32, MPAD, laneLo, laneHi);
        v16bf b = frag_rowK_f32(sh_KVT, et * 16, kt * 32, MPAD, laneLo, laneHi);
        acc = wmma_bf16(a, b, acc);
      }
#pragma unroll
      for (int kt = 0; kt < 2; ++kt) {   // intra-chunk: K over c=64 (vT staged)
        v16bf a = frag_rowK_bf(sh_A, mtO * 16, kt * 32, SPAD, laneLo, laneHi);
        v16bf b = frag_rowK_bf(stage, et * 16, kt * 32, SPAD, laneLo, laneHi);
        acc = wmma_bf16(a, b, acc);
      }
      accO[tt] = acc;
    }
    __syncthreads();   // denom published; all reads of old sh_KVT done

    // ---- scale + store output ------------------------------------------
#pragma unroll
    for (int tt = 0; tt < 2; ++tt) {
      int et = (w & 1) * 2 + tt;
#pragma unroll
      for (int r = 0; r < 8; ++r) {
        int row = mtO * 16 + laneHi * 8 + r;
        outg[cb + (size_t)row * 64 + et * 16 + laneLo] = accO[tt][r] * denomv[row];
      }
    }

    // ---- KV += kpT @ v, stored transposed: C rows contiguous per lane ---
#pragma unroll
    for (int mi = 0; mi < 2; ++mi) {
      int mbase = (2 * w + mi) * 16;
      v16bf a0 = frag_colK_bf(sh_kp, 0,  mbase, MPAD, laneLo, laneHi);
      v16bf a1 = frag_colK_bf(sh_kp, 32, mbase, MPAD, laneLo, laneHi);
#pragma unroll
      for (int et = 0; et < 4; ++et) {
        v16bf b0 = frag_rowK_bf(stage, et * 16, 0,  SPAD, laneLo, laneHi);
        v16bf b1 = frag_rowK_bf(stage, et * 16, 32, SPAD, laneLo, laneHi);
        float* kv = sh_KVT + (et * 16 + laneLo) * MPAD + mbase + laneHi * 8;
        v8f c;
#pragma unroll
        for (int r = 0; r < 8; ++r) c[r] = kv[r];     // contiguous -> b128
        c = wmma_bf16(a0, b0, c);
        c = wmma_bf16(a1, b1, c);
#pragma unroll
        for (int r = 0; r < 8; ++r) kv[r] = c[r];     // contiguous -> b128
      }
    }
    // ---- k_state += colsum(kp) from register partials -------------------
    {
      float t0 = csum0 + __shfl_xor(csum0, 16, 32);
      float t1 = csum1 + __shfl_xor(csum1, 16, 32);
      if (laneHi == 0) {
        kstate[w * 32 + laneLo]      += t0;
        kstate[w * 32 + 16 + laneLo] += t1;
      }
    }
    __syncthreads();
  }
}

// ---------------------------------------------------------------------------
extern "C" void kernel_launch(void* const* d_in, const int* in_sizes, int n_in,
                              void* d_out, int out_size, void* d_ws, size_t ws_size,
                              hipStream_t stream) {
  (void)in_sizes; (void)n_in; (void)out_size; (void)ws_size;
  const float* q    = (const float*)d_in[0];
  const float* k    = (const float*)d_in[1];
  const float* v    = (const float*)d_in[2];
  const float* proj = (const float*)d_in[3];
  float* out = (float*)d_out;
  float* wsf = (float*)d_ws;   // wsf[0]=global k max, wsf[1..4096]=partials

  kmax_partial<<<4096, 256, 0, stream>>>(k, proj, wsf + 1);
  kmax_reduce<<<1, 256, 0, stream>>>(wsf + 1, wsf);

  (void)hipFuncSetAttribute((const void*)perf_causal_main,
                            hipFuncAttributeMaxDynamicSharedMemorySize,
                            (int)SMEM_BYTES);
  perf_causal_main<<<64, 256, SMEM_BYTES, stream>>>(q, k, v, proj, wsf, out);
}
